// BitLinear_45603962749242
// MI455X (gfx1250) — compile-verified, matrix operand order checked by reading, not run on hardware
//
#include <hip/hip_runtime.h>
#include <stdint.h>

typedef int v8i __attribute__((ext_vector_type(8)));

#define D_DIM     1024          // D_IN == D_OUT == 1024
#define NTOK      32768         // 4 * 8192 tokens
#define M_TILE    128           // token rows per workgroup
#define ROW_STRIDE 1040         // 1024 + 16B pad -> conflict-free ds_load_b64 frags
#define LDS_BYTES (M_TILE*ROW_STRIDE + M_TILE*4)

// -------------------------------------------------------------------------
// Kernel 1: deterministic partial sums of |w| (1024 blocks x 1024 elements)
// -------------------------------------------------------------------------
__global__ void k_wabs_partial(const float* __restrict__ w,
                               float* __restrict__ partial) {
    __shared__ float red[256];
    const int tid = threadIdx.x;
    const size_t base = (size_t)blockIdx.x * 1024 + tid;
    float s = 0.f;
#pragma unroll
    for (int i = 0; i < 4; ++i) s += fabsf(w[base + i * 256]);
    red[tid] = s;
    __syncthreads();
    for (int off = 128; off > 0; off >>= 1) {
        if (tid < off) red[tid] += red[tid + off];
        __syncthreads();
    }
    if (tid == 0) partial[blockIdx.x] = red[0];
}

// -------------------------------------------------------------------------
// Kernel 2: every block redundantly reduces the 1024 partials (deterministic,
// identical result), then quantizes its 1024 weights to ternary int8.
// -------------------------------------------------------------------------
__global__ void k_wquant(const float* __restrict__ w,
                         const float* __restrict__ partial,
                         float* __restrict__ wscale_out,
                         signed char* __restrict__ wq) {
    __shared__ float red[256];
    __shared__ float s_scale;
    const int tid = threadIdx.x;
    float s = 0.f;
#pragma unroll
    for (int i = 0; i < 4; ++i) s += partial[tid + i * 256];
    red[tid] = s;
    __syncthreads();
    for (int off = 128; off > 0; off >>= 1) {
        if (tid < off) red[tid] += red[tid + off];
        __syncthreads();
    }
    if (tid == 0) {
        float sc = fmaxf(red[0] * (1.0f / 1048576.0f), 1e-5f);
        s_scale = sc;
        if (blockIdx.x == 0) *wscale_out = sc;
    }
    __syncthreads();
    const float inv = 1.0f / s_scale;
    const size_t base = (size_t)blockIdx.x * 1024 + tid;
#pragma unroll
    for (int i = 0; i < 4; ++i) {
        float q = rintf(w[base + i * 256] * inv);
        q = fminf(fmaxf(q, -1.f), 1.f);
        wq[base + i * 256] = (signed char)(int)q;
    }
}

// -------------------------------------------------------------------------
// 8-bit 16x64 fragment load (A-layout; B assumed mirror with lane -> column):
// per lane four 8-byte chunks at kbase + 8*half + {0,16,32,48}.
// Works for LDS (ds_load_b64) and global (global_load_b64) pointers.
// -------------------------------------------------------------------------
__device__ __forceinline__ v8i load_frag8(const signed char* p) {
    v8i f;
    long long* q = (long long*)&f;
    q[0] = *(const long long*)(p + 0);
    q[1] = *(const long long*)(p + 16);
    q[2] = *(const long long*)(p + 32);
    q[3] = *(const long long*)(p + 48);
    return f;
}

__device__ __forceinline__ void store_tile(v8i acc, float* __restrict__ out,
                                           const float* __restrict__ sxl,
                                           float sw, int m0, int lrow0,
                                           int nbase, int half, int l16) {
#pragma unroll
    for (int j = 0; j < 8; ++j) {
        const int lrow = lrow0 + 8 * half + j;   // local row in tile
        out[(size_t)(m0 + lrow) * D_DIM + nbase + l16] =
            (float)acc[j] * sxl[lrow] * sw;
    }
}

// -------------------------------------------------------------------------
// Kernel 3: fused activation-quant + int8 WMMA GEMM.
// 256 threads = 8 waves; waves = 2 M-groups (64 rows) x 4 N-groups (256 cols).
// -------------------------------------------------------------------------
__global__ void __launch_bounds__(256)
k_gemm(const float* __restrict__ x, const signed char* __restrict__ wq,
       const float* __restrict__ wscale, float* __restrict__ out) {
    extern __shared__ __align__(16) signed char smem[];
    float* sxl = (float*)(smem + M_TILE * ROW_STRIDE);   // per-row scale/127

    const int tid  = threadIdx.x;
    const int lane = tid & 31;
    const int wv   = tid >> 5;
    const int m0   = blockIdx.x * M_TILE;
    const float sw = *wscale;

    // ---------------- Phase 1: quantize x tile into LDS -----------------
    for (int r = 0; r < 16; ++r) {
        const int row = wv * 16 + r;                       // 0..127
        const float4* px = (const float4*)(x + (size_t)(m0 + row) * D_DIM);
        float4 v[8];
        float amax = 0.f;
#pragma unroll
        for (int it = 0; it < 8; ++it) {
            float4 t = px[it * 32 + lane];
            v[it] = t;
            amax = fmaxf(amax, fmaxf(fmaxf(fabsf(t.x), fabsf(t.y)),
                                     fmaxf(fabsf(t.z), fabsf(t.w))));
        }
#pragma unroll
        for (int off = 16; off > 0; off >>= 1)
            amax = fmaxf(amax, __shfl_xor(amax, off, 32));
        const float scale = fmaxf(amax, 1e-5f);
        const float inv = 1.0f / scale;
#pragma unroll
        for (int it = 0; it < 8; ++it) {
            int q0 = (int)rintf(fminf(fmaxf(v[it].x * inv, -128.f), 127.f));
            int q1 = (int)rintf(fminf(fmaxf(v[it].y * inv, -128.f), 127.f));
            int q2 = (int)rintf(fminf(fmaxf(v[it].z * inv, -128.f), 127.f));
            int q3 = (int)rintf(fminf(fmaxf(v[it].w * inv, -128.f), 127.f));
            unsigned pk = (q0 & 255) | ((q1 & 255) << 8) |
                          ((q2 & 255) << 16) | ((unsigned)(q3 & 255) << 24);
            *(unsigned*)(smem + (size_t)row * ROW_STRIDE + it * 128 + lane * 4) = pk;
        }
        if (lane == 0) sxl[row] = scale * (1.0f / 127.0f);
    }
    __syncthreads();

    // ---------------- Phase 2: int8 WMMA GEMM ----------------------------
    const int half = lane >> 4;
    const int l16  = lane & 15;
    const int mg   = wv >> 2;    // 0..1 : rows [mg*64, mg*64+64)
    const int ng   = wv & 3;     // 0..3 : cols [ng*256, ng*256+256)

    const signed char* abase =
        smem + (size_t)(mg * 64 + l16) * ROW_STRIDE + 8 * half;

    for (int nt = 0; nt < 16; ++nt) {
        const int nbase = ng * 256 + nt * 16;
        const signed char* bp =
            wq + (size_t)(nbase + l16) * D_DIM + 8 * half;

        v8i acc0 = {0,0,0,0,0,0,0,0};
        v8i acc1 = {0,0,0,0,0,0,0,0};
        v8i acc2 = {0,0,0,0,0,0,0,0};
        v8i acc3 = {0,0,0,0,0,0,0,0};

        for (int k = 0; k < D_DIM; k += 64) {
            v8i b  = load_frag8(bp + k);
            v8i a0 = load_frag8(abase + 0 * 16 * ROW_STRIDE + k);
            v8i a1 = load_frag8(abase + 1 * 16 * ROW_STRIDE + k);
            v8i a2 = load_frag8(abase + 2 * 16 * ROW_STRIDE + k);
            v8i a3 = load_frag8(abase + 3 * 16 * ROW_STRIDE + k);
            acc0 = __builtin_amdgcn_wmma_i32_16x16x64_iu8(true, a0, true, b, acc0, false, false);
            acc1 = __builtin_amdgcn_wmma_i32_16x16x64_iu8(true, a1, true, b, acc1, false, false);
            acc2 = __builtin_amdgcn_wmma_i32_16x16x64_iu8(true, a2, true, b, acc2, false, false);
            acc3 = __builtin_amdgcn_wmma_i32_16x16x64_iu8(true, a3, true, b, acc3, false, false);
        }

        store_tile(acc0, out, sxl, sw, m0, mg * 64 + 0 * 16, nbase, half, l16);
        store_tile(acc1, out, sxl, sw, m0, mg * 64 + 1 * 16, nbase, half, l16);
        store_tile(acc2, out, sxl, sw, m0, mg * 64 + 2 * 16, nbase, half, l16);
        store_tile(acc3, out, sxl, sw, m0, mg * 64 + 3 * 16, nbase, half, l16);
    }
}

// -------------------------------------------------------------------------
extern "C" void kernel_launch(void* const* d_in, const int* in_sizes, int n_in,
                              void* d_out, int out_size, void* d_ws, size_t ws_size,
                              hipStream_t stream) {
    const float* x = (const float*)d_in[0];   // [4,8192,1024] fp32
    const float* w = (const float*)d_in[1];   // [1024,1024]   fp32
    float* out = (float*)d_out;               // [4,8192,1024] fp32

    float* partial      = (float*)d_ws;                       // 1024 floats
    float* wscale       = (float*)((char*)d_ws + 4096);       // 1 float
    signed char* wq     = (signed char*)d_ws + 8192;          // 1 MiB int8

    k_wabs_partial<<<1024, 256, 0, stream>>>(w, partial);
    k_wquant<<<1024, 256, 0, stream>>>(w, partial, wscale, wq);

    (void)hipFuncSetAttribute(reinterpret_cast<const void*>(k_gemm),
                              hipFuncAttributeMaxDynamicSharedMemorySize,
                              LDS_BYTES);
    k_gemm<<<NTOK / M_TILE, 256, LDS_BYTES, stream>>>(x, wq, wscale, out);
}
